// LocalWindowAttention_2628519985211
// MI455X (gfx1250) — matrix operand order repaired and checked by reading, not compiled, not run on hardware
//
#include <hip/hip_runtime.h>

// MI455X (gfx1250) sliding-window attention, FlashAttention-style.
// bf16 WMMA (v_wmma_f32_16x16x32_bf16) with f32 accumulation, wave32.
// One 128-thread block (4 waves) owns 64 query rows of one head; each wave
// owns 16 rows. Computes S^T = K*Q^T so the C-layout of S^T is directly the
// A-layout of P for the P*V WMMA (no cross-lane transpose needed).
// 1/sqrt(D) is folded into the Q staging; interior blocks skip masking;
// waves skip blocks fully outside their window; global tile loads are
// hoisted above the barrier to overlap the previous iteration's compute.

typedef __attribute__((ext_vector_type(16))) __bf16 v16bf;
typedef __attribute__((ext_vector_type(8)))  float  v8f;

#define WINDOW   256
#define S_LEN    4096
#define NHEADS   8
#define DHEAD    128
#define BM       64      // query rows per workgroup
#define BK       32      // key rows per iteration
#define NTHREADS 128

// pack two f32 into two bf16 (RNE via hardware convert)
__device__ __forceinline__ unsigned pack_bf16(float a, float b) {
  __bf16 x = (__bf16)a;
  __bf16 y = (__bf16)b;
  unsigned lo = (unsigned)__builtin_bit_cast(unsigned short, x);
  unsigned hi = (unsigned)__builtin_bit_cast(unsigned short, y);
  return lo | (hi << 16);
}

union V16 {
  v16bf v;
  uint4 q[2];
  unsigned u32[8];
};

__global__ __launch_bounds__(NTHREADS)
void lwa_wmma_kernel(const float* __restrict__ qg,
                     const float* __restrict__ kg,
                     const float* __restrict__ vg,
                     float* __restrict__ og) {
  __shared__ __align__(16) unsigned short sQ[BM * DHEAD];   // row-major bf16, 16KB
  __shared__ __align__(16) unsigned short sK[BK * DHEAD];   // row-major bf16, 8KB
  __shared__ __align__(16) unsigned short sVt[DHEAD * BK];  // transposed [d][key], 8KB

  const int q0   = blockIdx.x * BM;
  const int h    = blockIdx.y;
  const int tid  = threadIdx.x;
  const int wave = tid >> 5;
  const int lane = tid & 31;
  const int hl   = lane >> 4;     // 16-lane half within wave32
  const int l16  = lane & 15;

  const float scale = 0.08838834764831845f;      // 1/sqrt(128), folded into Q
  const float NEG_INF = -__builtin_inff();

  const size_t headOff = (size_t)h * S_LEN * DHEAD;
  const float* qh = qg + headOff;
  const float* kh = kg + headOff;
  const float* vh = vg + headOff;
  float*       oh = og + headOff;

  // ---- stage Q tile (64x128) fp32 -> bf16 * scale, packed b128 stores ----
  {
    const int row = tid >> 1;
    const int cb  = (tid & 1) * 64;
    const float4* src = (const float4*)(qh + (size_t)(q0 + row) * DHEAD + cb);
    uint4* dst = (uint4*)(sQ + row * DHEAD + cb);
    #pragma unroll
    for (int j = 0; j < 8; ++j) {
      const float4 f0 = src[2 * j];
      const float4 f1 = src[2 * j + 1];
      uint4 u;
      u.x = pack_bf16(f0.x * scale, f0.y * scale);
      u.y = pack_bf16(f0.z * scale, f0.w * scale);
      u.z = pack_bf16(f1.x * scale, f1.y * scale);
      u.w = pack_bf16(f1.z * scale, f1.w * scale);
      dst[j] = u;
    }
  }
  __syncthreads();

  // ---- preload Q as WMMA B-operands (B 32x16 bf16 layout), 4 d-chunks ----
  // B layout: lanes 0-15 hold K=0..7,16..23; lanes 16-31 hold K=8..15,24..31 (K = d here).
  V16 qb[4];
  {
    const unsigned short* base = sQ + (wave * 16 + l16) * DHEAD;
    #pragma unroll
    for (int nc = 0; nc < 4; ++nc) {
      const uint4* p = (const uint4*)(base + nc * 32 + hl * 8);
      qb[nc].q[0] = p[0];
      qb[nc].q[1] = p[2];
    }
  }

  // ---- accumulators: out tile 16(q) x 128(d) = 8 C tiles ----
  v8f zero;
  #pragma unroll
  for (int j = 0; j < 8; ++j) zero[j] = 0.0f;
  v8f acc[8];
  #pragma unroll
  for (int nd = 0; nd < 8; ++nd) acc[nd] = zero;

  float m_r = -1e30f;   // running row max (finite sentinel)
  float l_r = 0.0f;     // running row sum
  const int qw     = q0 + wave * 16;   // wave's first query row
  const int qrow_g = qw + l16;         // query row this lane tracks

  int lo = q0 - (WINDOW - 1);
  if (lo < 0) lo = 0;
  lo &= ~(BK - 1);
  const int hi = q0 + BM - 1;

  // staging roles (fixed per thread)
  const int krow = tid >> 2;            // K row 0..31
  const int kcb  = (tid & 3) * 32;      // K col base
  const int vkr  = (tid & 15) * 2;      // V even key pair
  const int vds  = (tid >> 4) * 16;     // V d-segment base

  for (int kb = lo; kb <= hi; kb += BK) {
    // ---- issue global tile loads BEFORE the barrier (overlap prev compute) ----
    float4 kf[8];
    {
      const float4* ks = (const float4*)(kh + (size_t)(kb + krow) * DHEAD + kcb);
      #pragma unroll
      for (int j = 0; j < 8; ++j) kf[j] = ks[j];
    }
    float4 vfa[4], vfb[4];
    {
      const float4* va = (const float4*)(vh + (size_t)(kb + vkr)     * DHEAD + vds);
      const float4* vb = (const float4*)(vh + (size_t)(kb + vkr + 1) * DHEAD + vds);
      #pragma unroll
      for (int j = 0; j < 4; ++j) { vfa[j] = va[j]; vfb[j] = vb[j]; }
    }

    __syncthreads();   // previous iteration's LDS reads done before overwrite

    // ---- K (32x128) row-major, packed b128 stores ----
    {
      uint4* kd = (uint4*)(sK + krow * DHEAD + kcb);
      #pragma unroll
      for (int j = 0; j < 4; ++j) {
        const float4 f0 = kf[2 * j];
        const float4 f1 = kf[2 * j + 1];
        uint4 u;
        u.x = pack_bf16(f0.x, f0.y);
        u.y = pack_bf16(f0.z, f0.w);
        u.z = pack_bf16(f1.x, f1.y);
        u.w = pack_bf16(f1.z, f1.w);
        kd[j] = u;
      }
    }
    // ---- V transposed [d][key]: two adjacent keys -> one b32 store ----
    {
      #pragma unroll
      for (int j = 0; j < 4; ++j) {
        const float4 a = vfa[j];
        const float4 b = vfb[j];
        const int d = vds + j * 4;
        *(unsigned*)(sVt + (d + 0) * BK + vkr) = pack_bf16(a.x, b.x);
        *(unsigned*)(sVt + (d + 1) * BK + vkr) = pack_bf16(a.y, b.y);
        *(unsigned*)(sVt + (d + 2) * BK + vkr) = pack_bf16(a.z, b.z);
        *(unsigned*)(sVt + (d + 3) * BK + vkr) = pack_bf16(a.w, b.w);
      }
    }
    __syncthreads();

    // ---- wave-uniform skip: block fully outside this wave's window? ----
    const bool waveActive = (kb <= qw + 15) && (kb + BK - 1 >= qw - (WINDOW - 1));
    if (waveActive) {
      // ---- S^T = K * Q^T : two 16x16 C tiles (keys 0-15, 16-31), K-dim 128 ----
      v8f c0 = zero, c1 = zero;
      #pragma unroll
      for (int nc = 0; nc < 4; ++nc) {
        // A 16x32 bf16 layout: lanes 0-15: K=0..7 (V0-3), K=16..23 (V4-7);
        //                      lanes 16-31: K=8..15, K=24..31.
        V16 a0, a1;
        const unsigned short* kb0 = sK + l16 * DHEAD + nc * 32 + hl * 8;
        a0.q[0] = *(const uint4*)(kb0);
        a0.q[1] = *(const uint4*)(kb0 + 16);
        const unsigned short* kb1 = sK + (16 + l16) * DHEAD + nc * 32 + hl * 8;
        a1.q[0] = *(const uint4*)(kb1);
        a1.q[1] = *(const uint4*)(kb1 + 16);
        c0 = __builtin_amdgcn_wmma_f32_16x16x32_bf16(false, a0.v, false, qb[nc].v,
                                                     (short)0, c0, false, false);
        c1 = __builtin_amdgcn_wmma_f32_16x16x32_bf16(false, a1.v, false, qb[nc].v,
                                                     (short)0, c1, false, false);
      }

      // ---- mask (only on edge blocks) + block max ----
      // C layout of S^T: VGPR i -> key = kb + tile*16 + hl*8 + i, query = qrow_g.
      const bool needMask = (kb + BK - 1 > qw) || (kb < qw + 15 - (WINDOW - 1));
      float bmax = NEG_INF;
      if (!needMask) {
        #pragma unroll
        for (int i = 0; i < 8; ++i) bmax = fmaxf(bmax, fmaxf(c0[i], c1[i]));
      } else {
        #pragma unroll
        for (int i = 0; i < 8; ++i) {
          const int key0 = kb + hl * 8 + i;
          const int key1 = key0 + 16;
          float s0 = c0[i];
          float s1 = c1[i];
          s0 = ((key0 <= qrow_g) && (key0 + WINDOW > qrow_g)) ? s0 : NEG_INF;
          s1 = ((key1 <= qrow_g) && (key1 + WINDOW > qrow_g)) ? s1 : NEG_INF;
          c0[i] = s0; c1[i] = s1;
          bmax = fmaxf(bmax, fmaxf(s0, s1));
        }
      }
      bmax = fmaxf(bmax, __shfl_xor(bmax, 16, 32));
      const float m_new = fmaxf(m_r, bmax);
      const float fcorr = __expf(m_r - m_new);  // ==1 if block fully masked, ==0 first time

      float p[16];
      float rs = 0.0f;
      #pragma unroll
      for (int i = 0; i < 8; ++i) {
        const float p0 = __expf(c0[i] - m_new); // masked: exp(-inf) = 0
        const float p1 = __expf(c1[i] - m_new);
        p[i] = p0; p[8 + i] = p1;
        rs += p0 + p1;
      }
      rs += __shfl_xor(rs, 16, 32);
      l_r = l_r * fcorr + rs;
      m_r = m_new;

      // ---- rescale accumulators: factor for row r held by lane r ----
      #pragma unroll
      for (int i = 0; i < 8; ++i) {
        const float fi = __shfl(fcorr, hl * 8 + i, 32);
        #pragma unroll
        for (int nd = 0; nd < 8; ++nd) acc[nd][i] *= fi;
      }

      // ---- P (16q x 32k) A-operand: direct per-lane pack of S^T C tiles ----
      V16 ap;
      #pragma unroll
      for (int j = 0; j < 4; ++j) {
        ap.u32[j]     = pack_bf16(p[2 * j],     p[2 * j + 1]);
        ap.u32[4 + j] = pack_bf16(p[8 + 2 * j], p[8 + 2 * j + 1]);
      }

      // ---- acc += P * V : 8 N-tiles across D=128, B from transposed V ----
      #pragma unroll
      for (int nd = 0; nd < 8; ++nd) {
        V16 bv;
        const unsigned short* vb = sVt + (nd * 16 + l16) * BK + hl * 8;
        bv.q[0] = *(const uint4*)(vb);
        bv.q[1] = *(const uint4*)(vb + 16);
        acc[nd] = __builtin_amdgcn_wmma_f32_16x16x32_bf16(false, ap.v, false, bv.v,
                                                          (short)0, acc[nd], false, false);
      }
    }
  }

  // ---- normalize and write out (C layout: VGPR i -> row i + 8*hl, col = nd*16+l16) ----
  #pragma unroll
  for (int i = 0; i < 8; ++i) {
    const float li   = __shfl(l_r, hl * 8 + i, 32);
    const float linv = 1.0f / li;
    const int row = q0 + wave * 16 + hl * 8 + i;
    float* orow = oh + (size_t)row * DHEAD;
    #pragma unroll
    for (int nd = 0; nd < 8; ++nd) {
      orow[nd * 16 + l16] = acc[nd][i] * linv;
    }
  }
}

extern "C" void kernel_launch(void* const* d_in, const int* in_sizes, int n_in,
                              void* d_out, int out_size, void* d_ws, size_t ws_size,
                              hipStream_t stream) {
  const float* q = (const float*)d_in[0];
  const float* k = (const float*)d_in[1];
  const float* v = (const float*)d_in[2];
  float* out = (float*)d_out;
  dim3 grid(S_LEN / BM, NHEADS);
  lwa_wmma_kernel<<<grid, NTHREADS, 0, stream>>>(q, k, v, out);
}
